// PACE_19567871000638
// MI455X (gfx1250) — matrix operand order repaired and checked by reading, not compiled, not run on hardware
//
#include <hip/hip_runtime.h>
#include <math.h>

// ---------------- types ----------------
typedef __bf16 bf16;
typedef __attribute__((ext_vector_type(8)))  __bf16 bf16x8;
typedef __attribute__((ext_vector_type(16))) __bf16 bf16x16;
typedef __attribute__((ext_vector_type(8)))  float  f32x8;

// ---------------- problem constants ----------------
#define G_     2048
#define NPG_   64
#define EPG_   128
#define ET_    (G_*EPG_)
#define DD_    78
#define DN_    200
#define KSEL_  52          // ceil(0.8*64)
#define M1_    4096        // x1;x2 stacked
#define N1_    1024
#define K1_    8000
#define K2_    1024
#define N2P_   112         // 100 padded to 7 tiles
#define KD_    160         // 156 padded
#define ND_    208         // 200 padded

// ---------------- workspace layout (bytes) ----------------
#define WS_XB     0UL                  // bf16 [4096][8000]  = 65,536,000
#define WS_W1S    65536000UL           // bf16 [1024][8000]  = 16,384,000
#define WS_H1     81920000UL           // bf16 [4096][1024]  =  8,388,608
#define WS_BIAS1  90308608UL           // f32  [1024]
#define WS_W2B    90312704UL           // bf16 [112][1024]
#define WS_WCAT   90542080UL           // bf16 [208][160]
#define WS_DX     90608640UL           // f32  [2048][400]   =  3,276,800
// total ~93.9 MB

// ---------------- WMMA helpers (CDNA5 wave32 layouts) ----------------
// A fragment (16x32 bf16, M x K): lane m=lane&15, half=lane>>4.
// elements 0..7  <-> K = k0 + 8*half + e
// elements 8..15 <-> K = k0 + 16 + 8*half + (e-8)
static __device__ inline bf16x16 frag_a(const bf16* p) {
  bf16x8 lo = *(const bf16x8*)(p);
  bf16x8 hi = *(const bf16x8*)(p + 16);
  bf16x16 r;
#pragma unroll
  for (int i = 0; i < 8; ++i) { r[i] = lo[i]; r[i + 8] = hi[i]; }
  return r;
}
// B fragment (32x16 bf16, K x N): lane n=lane&15, half=lane>>4.
// elements 0..7 <-> K = k0 + 8*half + e; elements 8..15 <-> K = k0 + 16 + 8*half + (e-8).
static __device__ inline bf16x16 frag_b(const bf16* p) {
  bf16x8 lo = *(const bf16x8*)(p);
  bf16x8 hi = *(const bf16x8*)(p + 16);
  bf16x16 r;
#pragma unroll
  for (int i = 0; i < 8; ++i) { r[i] = lo[i]; r[i + 8] = hi[i]; }
  return r;
}
static __device__ inline f32x8 wmma_bf16(bf16x16 a, bf16x16 b, f32x8 c) {
  return __builtin_amdgcn_wmma_f32_16x16x32_bf16(false, a, false, b, (short)0, c,
                                                 false, false);
}

// ---------------- prep kernels ----------------
__global__ __launch_bounds__(256) void conv_x_kernel(const float* __restrict__ x1,
                                                     const float* __restrict__ x2,
                                                     bf16* __restrict__ Xb) {
  const size_t half = 2048UL * 8000UL, tot = 2 * half;
  for (size_t i = (size_t)blockIdx.x * 256 + threadIdx.x; i < tot;
       i += (size_t)gridDim.x * 256)
    Xb[i] = (bf16)(i < half ? x1[i] : x2[i - half]);
}

__global__ __launch_bounds__(256) void prep_w1_kernel(const float* __restrict__ W1,
                                                      const float* __restrict__ s0,
                                                      bf16* __restrict__ W1s) {
  const size_t tot = 1024UL * 8000UL;
  for (size_t i = (size_t)blockIdx.x * 256 + threadIdx.x; i < tot;
       i += (size_t)gridDim.x * 256)
    W1s[i] = (bf16)(W1[i] * s0[i % 8000]);
}

// bias1[j] = b1[j] + sum_k W1[j,k] * bn0_t[k]
__global__ __launch_bounds__(256) void prep_bias1_kernel(const float* __restrict__ W1,
                                                         const float* __restrict__ t0,
                                                         const float* __restrict__ b1,
                                                         float* __restrict__ bias1) {
  int j = blockIdx.x * 256 + threadIdx.x;
  if (j >= 1024) return;
  float acc = b1[j];
  const float* row = W1 + (size_t)j * 8000;
  for (int k = 0; k < 8000; ++k) acc += row[k] * t0[k];
  bias1[j] = acc;
}

__global__ __launch_bounds__(256) void prep_w2_kernel(const float* __restrict__ W2,
                                                      bf16* __restrict__ W2b) {
  int i = blockIdx.x * 256 + threadIdx.x;
  if (i >= N2P_ * K2_) return;
  int n = i / K2_, k = i % K2_;
  W2b[i] = (bf16)(n < 100 ? W2[n * K2_ + k] : 0.f);
}

// Wcat[n][c]: c<78 -> Wrel[n][c]; 78<=c<156 -> Wroot[n][c-78]; else 0; rows>=200 zero
__global__ __launch_bounds__(256) void prep_wcat_kernel(const float* __restrict__ Wrel,
                                                        const float* __restrict__ Wroot,
                                                        bf16* __restrict__ Wcat) {
  int i = blockIdx.x * 256 + threadIdx.x;
  if (i >= ND_ * KD_) return;
  int n = i / KD_, c = i % KD_;
  float v = 0.f;
  if (n < DN_) {
    if (c < DD_) v = Wrel[n * DD_ + c];
    else if (c < 2 * DD_) v = Wroot[n * DD_ + (c - DD_)];
  }
  Wcat[i] = (bf16)v;
}

// ---------------- GEMM1: [4096,8000]bf16 x [8000,1024]bf16 -> BN1+ReLU -> bf16 H1 ----
// Wave tile 64(M)x32(N): 8 WMMAs per 12 x 16B loads per K=32 step (~21 flop/B).
// Block: 8 waves = 4 M-waves x 2 N-waves -> block tile 256(M) x 64(N).
__global__ __launch_bounds__(256) void gemm1_kernel(const bf16* __restrict__ Xb,
                                                    const bf16* __restrict__ W1s,
                                                    const float* __restrict__ bias1,
                                                    const float* __restrict__ s1,
                                                    const float* __restrict__ t1,
                                                    bf16* __restrict__ H1) {
  int lane = threadIdx.x & 31, wave = threadIdx.x >> 5;
  int lm = lane & 15, hs = lane >> 4;
  int m0 = blockIdx.y * 256 + (wave & 3) * 64;   // wave rows m0..m0+63
  int n0 = blockIdx.x * 64 + (wave >> 2) * 32;   // wave cols n0..n0+31

  const bf16* aR[4];
#pragma unroll
  for (int i = 0; i < 4; ++i)
    aR[i] = Xb + (size_t)(m0 + i * 16 + lm) * K1_ + 8 * hs;
  const bf16* bR0 = W1s + (size_t)(n0 + lm)      * K1_ + 8 * hs;
  const bf16* bR1 = W1s + (size_t)(n0 + 16 + lm) * K1_ + 8 * hs;

  f32x8 acc[4][2] = {};
  for (int k0 = 0; k0 < K1_; k0 += 32) {
    bf16x16 b0 = frag_b(bR0 + k0);
    bf16x16 b1 = frag_b(bR1 + k0);
#pragma unroll
    for (int i = 0; i < 4; ++i) {
      bf16x16 a = frag_a(aR[i] + k0);
      acc[i][0] = wmma_bf16(a, b0, acc[i][0]);
      acc[i][1] = wmma_bf16(a, b1, acc[i][1]);
    }
  }
#pragma unroll
  for (int j = 0; j < 2; ++j) {
    int n = n0 + j * 16 + lm;
    float bb = bias1[n], ss = s1[n], tt = t1[n];
#pragma unroll
    for (int i = 0; i < 4; ++i) {
#pragma unroll
      for (int v = 0; v < 8; ++v) {
        int m = m0 + i * 16 + v + 8 * hs;   // C layout: M = v + 8*half, N = lane&15
        float y = (acc[i][j][v] + bb) * ss + tt;
        y = y > 0.f ? y : 0.f;
        H1[(size_t)m * N1_ + n] = (bf16)y;
      }
    }
  }
}

// ---------------- GEMM2: H1 x W2b^T -> BN2+ReLU -> f1/f2 in d_out ----------------
__global__ __launch_bounds__(256) void gemm2_kernel(const bf16* __restrict__ H1,
                                                    const bf16* __restrict__ W2b,
                                                    const float* __restrict__ b2,
                                                    const float* __restrict__ s2,
                                                    const float* __restrict__ t2,
                                                    float* __restrict__ out) {
  int lane = threadIdx.x & 31, wave = threadIdx.x >> 5;
  int lm = lane & 15, hs = lane >> 4;
  int m0 = blockIdx.x * 128 + wave * 16;

  const bf16* aR = H1 + (size_t)(m0 + lm) * K2_ + 8 * hs;
  f32x8 acc[7] = {};
  for (int k0 = 0; k0 < K2_; k0 += 32) {
    bf16x16 a = frag_a(aR + k0);
#pragma unroll
    for (int j = 0; j < 7; ++j) {
      const bf16* bR = W2b + (size_t)(j * 16 + lm) * K2_ + 8 * hs + k0;
      acc[j] = wmma_bf16(a, frag_b(bR), acc[j]);
    }
  }
#pragma unroll
  for (int j = 0; j < 7; ++j) {
    int n = j * 16 + lm;
    if (n < 100) {
      float bb = b2[n], ss = s2[n], tt = t2[n];
#pragma unroll
      for (int v = 0; v < 8; ++v) {
        int m = m0 + v + 8 * hs;
        float y = (acc[j][v] + bb) * ss + tt;
        y = y > 0.f ? y : 0.f;
        if (m < 2048) out[(size_t)m * 100 + n] = y;
        else          out[204800 + (size_t)(m - 2048) * 100 + n] = y;
      }
    }
  }
}

// ---------------- drug encoder: one block per graph ----------------
__global__ __launch_bounds__(256) void drug_kernel(const float* __restrict__ drug_x,
                                                   const int* __restrict__ edge_index,
                                                   const bf16* __restrict__ Wcat,
                                                   const float* __restrict__ brel,
                                                   const float* __restrict__ bnd_s,
                                                   const float* __restrict__ bnd_t,
                                                   const float* __restrict__ topk_w,
                                                   float* __restrict__ dxout) {
  __shared__ __align__(16) char smem[62208];
  float* xs   = (float*)smem;                 // [64*80] f32       (phase 1-3)
  float* ag   = (float*)(smem + 20480);       // [64*80] f32       (phase 1-3)
  bf16*  a_bf = (bf16*)(smem + 40960);        // [64*160] bf16     (phase 3-4)
  bf16*  h_sh = (bf16*)smem;                  // [64*208] bf16     (phase 4+, aliases xs/ag)
  float* s_sh = (float*)(smem + 61440);       // [64]
  int*   sel  = (int*)(smem + 61696);         // [64]
  float* wn   = (float*)(smem + 61952);       // [1]

  int g = blockIdx.x, tid = threadIdx.x;

  // phase 1: load node features, zero aggregation
  for (int i = tid; i < NPG_ * DD_; i += 256) {
    int node = i / DD_, f = i % DD_;
    xs[node * 80 + f] = drug_x[(size_t)(g * NPG_ + node) * DD_ + f];
  }
  for (int i = tid; i < NPG_ * 80; i += 256) ag[i] = 0.f;
  __syncthreads();

  // phase 2: segment-sum over this graph's 128 edges (LDS atomics)
  {
    int e = tid & 127, half = tid >> 7;
    int eg = g * EPG_ + e;
    int ls = edge_index[eg]       - g * NPG_;
    int ld = edge_index[ET_ + eg] - g * NPG_;
    int f0 = half * 39, f1 = f0 + 39;
    for (int f = f0; f < f1; ++f) atomicAdd(&ag[ld * 80 + f], xs[ls * 80 + f]);
  }
  __syncthreads();

  // phase 3: build bf16 A tile [64][160] = [agg | x | 0pad]
  for (int i = tid; i < NPG_ * KD_; i += 256) {
    int m = i / KD_, c = i % KD_;
    float v = (c < DD_) ? ag[m * 80 + c]
            : (c < 2 * DD_) ? xs[m * 80 + (c - DD_)] : 0.f;
    a_bf[i] = (bf16)v;
  }
  if (tid == 0) {
    float acc = 0.f;
    for (int j = 0; j < DN_; ++j) { float w = topk_w[j]; acc += w * w; }
    *wn = sqrtf(acc);
  }
  __syncthreads();

  // phase 4: WMMA  [64x160] @ [160x208]  -> BN+ReLU -> h_sh (bf16)
  int lane = tid & 31, wave = tid >> 5, lm = lane & 15, hs = lane >> 4;
  const bf16* aBase = a_bf + lm * KD_ + 8 * hs;
  for (int nt = wave; nt < 13; nt += 8) {      // wave-uniform
    f32x8 acc[4] = {};
    const bf16* bR = Wcat + (size_t)(nt * 16 + lm) * KD_ + 8 * hs;
#pragma unroll
    for (int k0 = 0; k0 < KD_; k0 += 32) {
      bf16x16 b = frag_b(bR + k0);
#pragma unroll
      for (int mt = 0; mt < 4; ++mt)
        acc[mt] = wmma_bf16(frag_a(aBase + mt * 16 * KD_ + k0), b, acc[mt]);
    }
    int n = nt * 16 + lm;
    if (n < DN_) {
      float bb = brel[n], ss = bnd_s[n], tt = bnd_t[n];
#pragma unroll
      for (int mt = 0; mt < 4; ++mt)
#pragma unroll
        for (int v = 0; v < 8; ++v) {
          int m = mt * 16 + v + 8 * hs;
          float y = (acc[mt][v] + bb) * ss + tt;
          y = y > 0.f ? y : 0.f;
          h_sh[m * ND_ + n] = (bf16)y;
        }
    }
  }
  __syncthreads();

  // phase 5: scores = tanh((h . w)/||w||)
  if (tid < NPG_) {
    float acc = 0.f;
    for (int c = 0; c < DN_; ++c) acc += (float)h_sh[tid * ND_ + c] * topk_w[c];
    s_sh[tid] = tanhf(acc / *wn);
  }
  __syncthreads();
  // rank-based top-K selection (K=52 of 64)
  if (tid < NPG_) {
    float si = s_sh[tid];
    int r = 0;
    for (int j = 0; j < NPG_; ++j) {
      float sj = s_sh[j];
      r += (sj > si) || (sj == si && j < tid);
    }
    sel[tid] = (r < KSEL_) ? 1 : 0;
  }
  __syncthreads();

  // phase 6: dx = relu(mm0 + 3*mm1), [max(200) | mean(200)]
  for (int c = tid; c < DN_; c += 256) {
    float m0v = -3.4e38f, m1v = -3.4e38f, sum0 = 0.f, sum1 = 0.f;
    for (int i = 0; i < NPG_; ++i) {
      float hv = (float)h_sh[i * ND_ + c];
      m0v = fmaxf(m0v, hv);
      sum0 += hv;
      if (sel[i]) {
        float xv = hv * s_sh[i];
        m1v = fmaxf(m1v, xv);
        sum1 += xv;
      }
    }
    float dmax  = m0v + 3.f * m1v;                       dmax  = dmax  > 0.f ? dmax  : 0.f;
    float dmean = sum0 * (1.f / 64.f) + 3.f * (sum1 * (1.f / 52.f));
    dmean = dmean > 0.f ? dmean : 0.f;
    dxout[(size_t)g * 400 + c]       = dmax;
    dxout[(size_t)g * 400 + 200 + c] = dmean;
  }
}

// ---------------- final: resp = [f1 | dx] @ Wp^T + bp ----------------
__global__ __launch_bounds__(256) void resp_kernel(const float* __restrict__ f1,
                                                   const float* __restrict__ dx,
                                                   const float* __restrict__ Wp,
                                                   const float* __restrict__ bp,
                                                   float* __restrict__ out) {
  int g = blockIdx.x * 256 + threadIdx.x;
  if (g >= 2048) return;
  float acc = bp[0];
  for (int j = 0; j < 100; ++j) acc += f1[(size_t)g * 100 + j] * Wp[j];
  for (int j = 0; j < 400; ++j) acc += dx[(size_t)g * 400 + j] * Wp[100 + j];
  out[409600 + g] = acc;
}

// ---------------- launcher ----------------
extern "C" void kernel_launch(void* const* d_in, const int* in_sizes, int n_in,
                              void* d_out, int out_size, void* d_ws, size_t ws_size,
                              hipStream_t stream) {
  (void)in_sizes; (void)n_in; (void)out_size; (void)ws_size;
  const float* x1     = (const float*)d_in[0];
  const float* x2     = (const float*)d_in[1];
  const int*   eidx   = (const int*)d_in[4];
  const float* drug_x = (const float*)d_in[5];
  const float* bn0_s  = (const float*)d_in[6];
  const float* bn0_t  = (const float*)d_in[7];
  const float* W1     = (const float*)d_in[8];
  const float* b1     = (const float*)d_in[9];
  const float* bn1_s  = (const float*)d_in[10];
  const float* bn1_t  = (const float*)d_in[11];
  const float* W2     = (const float*)d_in[12];
  const float* b2     = (const float*)d_in[13];
  const float* bn2_s  = (const float*)d_in[14];
  const float* bn2_t  = (const float*)d_in[15];
  const float* Wrel   = (const float*)d_in[16];
  const float* brel   = (const float*)d_in[17];
  const float* Wroot  = (const float*)d_in[18];
  const float* bnd_s  = (const float*)d_in[19];
  const float* bnd_t  = (const float*)d_in[20];
  const float* topk_w = (const float*)d_in[21];
  const float* Wp     = (const float*)d_in[22];
  const float* bp     = (const float*)d_in[23];
  float* out = (float*)d_out;

  char* ws = (char*)d_ws;
  bf16*  Xb    = (bf16*)(ws + WS_XB);
  bf16*  W1s   = (bf16*)(ws + WS_W1S);
  bf16*  H1    = (bf16*)(ws + WS_H1);
  float* bias1 = (float*)(ws + WS_BIAS1);
  bf16*  W2b   = (bf16*)(ws + WS_W2B);
  bf16*  Wcat  = (bf16*)(ws + WS_WCAT);
  float* dx    = (float*)(ws + WS_DX);

  conv_x_kernel   <<<4096, 256, 0, stream>>>(x1, x2, Xb);
  prep_w1_kernel  <<<2048, 256, 0, stream>>>(W1, bn0_s, W1s);
  prep_bias1_kernel<<<4,   256, 0, stream>>>(W1, bn0_t, b1, bias1);
  prep_w2_kernel  <<<448,  256, 0, stream>>>(W2, W2b);
  prep_wcat_kernel<<<130,  256, 0, stream>>>(Wrel, Wroot, Wcat);

  gemm1_kernel<<<dim3(N1_ / 64, M1_ / 256), 256, 0, stream>>>(Xb, W1s, bias1,
                                                              bn1_s, bn1_t, H1);
  gemm2_kernel<<<M1_ / 128, 256, 0, stream>>>(H1, W2b, b2, bn2_s, bn2_t, out);
  drug_kernel <<<G_, 256, 0, stream>>>(drug_x, eidx, Wcat, brel, bnd_s, bnd_t,
                                       topk_w, dx);
  resp_kernel <<<8, 256, 0, stream>>>(out, dx, Wp, bp, out);
}
